// Attention_20177756356854
// MI455X (gfx1250) — matrix operand order repaired and checked by reading, not compiled, not run on hardware
//
#include <hip/hip_runtime.h>
#include <hip/hip_bf16.h>
#include <math.h>

// ---------------- constants ----------------
constexpr int BATCH = 4;
constexpr int C_IN  = 192;
constexpr int C3    = 576;     // 3*C
constexpr int HEADS = 4;
constexpr int CH    = 48;      // channels per head
constexpr int HDIM  = 128;
constexpr int NTOK  = HDIM * HDIM;  // 16384 tokens
constexpr int KSPLIT = 8;

typedef __attribute__((ext_vector_type(16))) _Float16 v16h;
typedef __attribute__((ext_vector_type(8)))  _Float16 v8h;
typedef __attribute__((ext_vector_type(8)))  float    v8f;
typedef __attribute__((ext_vector_type(4)))  unsigned int u32x4;
typedef __attribute__((ext_vector_type(8)))  int          i32x8;
typedef __attribute__((ext_vector_type(4)))  int          i32x4;

union V16H { v16h v; _Float16 h[16]; };
union V8H  { v8h  v; _Float16 h[8];  };
union V8F  { v8f  v; float    f[8];  };

// ---------------- TDM availability ----------------
#if defined(__AMDGCN__) && __has_builtin(__builtin_amdgcn_tensor_load_to_lds) && \
    __has_builtin(__builtin_amdgcn_s_wait_tensorcnt)
#define HAVE_TDM 1
#else
#define HAVE_TDM 0
#endif

#if HAVE_TDM
// Issue a 2-D TDM load: tile_k rows x tile_n f16 elements, row stride
// row_stride (elements), from gptr (tile start) into LDS byte offset lds_off.
__device__ inline void tdm_load_2d(unsigned int lds_off, const void* gptr,
                                   int tile_n, int tile_k,
                                   unsigned int row_stride,
                                   unsigned int tdim0, unsigned int tdim1) {
    unsigned long long ga = (unsigned long long)(uintptr_t)gptr;
    // D# group0: count=1 | lds_addr | global_addr[56:0] | type=2
    u32x4 g0 = { 1u,
                 lds_off,
                 (unsigned int)ga,
                 (unsigned int)((ga >> 32) & 0x01FFFFFFu) | (2u << 30) };
    // D# group1: data_size=1 (2 bytes); tensor_dim0/1; tile_dim0/1; dim0 stride
    i32x8 g1 = { (int)(1u << 16),
                 (int)((tdim0 & 0xFFFFu) << 16),
                 (int)((tdim0 >> 16) | ((tdim1 & 0xFFFFu) << 16)),
                 (int)((tdim1 >> 16) | ((unsigned int)tile_n << 16)),
                 (int)tile_k,
                 (int)row_stride,
                 0, 0 };
    i32x4 z4 = {0, 0, 0, 0};
#if __clang_major__ >= 23
    i32x8 z8 = {0, 0, 0, 0, 0, 0, 0, 0};
    __builtin_amdgcn_tensor_load_to_lds(g0, g1, z4, z4, z8, 0);
#else
    __builtin_amdgcn_tensor_load_to_lds(g0, g1, z4, z4, 0);
#endif
}
#endif

// ---------------- WMMA helpers (wave32, 16x16x32 f16 -> f32) ----------------

__device__ inline v8f wmma_f32(v16h a, v16h b, v8f c) {
    return __builtin_amdgcn_wmma_f32_16x16x32_f16(false, a, false, b, (short)0, c,
                                                  false, false);
}

// A tile 16x32 (MxK), row-major A[M][lda], contiguous along K.
__device__ inline v16h load_a_tile(const _Float16* A, int lda, int mBase, int kBase,
                                   int lane) {
    int half = lane >> 4, l = lane & 15;
    const _Float16* row = A + (size_t)(mBase + l) * (size_t)lda;
    V16H a;
    *(v8h*)&a.h[0] = *(const v8h*)(row + kBase + half * 16);
    *(v8h*)&a.h[8] = *(const v8h*)(row + kBase + half * 16 + 8);
    return a.v;
}

// B tile where element (k, n) = Rows[n][k] (rows contiguous along k):
// one aligned 32B vector load per lane. Used for q·k^T.
__device__ inline v16h load_b_tile_rows(const _Float16* Rows, size_t ld, int nBase,
                                        int kBase, int lane) {
    int half = lane >> 4, l = lane & 15;
    const _Float16* p = Rows + (size_t)(nBase + l) * ld + kBase + half * 16;
    return *(const v16h*)p;
}

// B operand from an LDS panel laid out [32 k][panelW n] f16 (k-major).
__device__ inline v16h load_b_tile_lds(const _Float16* bp, int panelW, int nLocal,
                                       int lane) {
    int half = lane >> 4, l = lane & 15;
    V16H b;
#pragma unroll
    for (int e = 0; e < 16; ++e)
        b.h[e] = bp[(half * 16 + e) * panelW + nLocal + l];
    return b.v;
}

// ---------------- utility kernels ----------------

__global__ void cvt_f32_to_f16(const float* __restrict__ src, _Float16* __restrict__ dst,
                               size_t n) {
    size_t i = (size_t)blockIdx.x * blockDim.x + threadIdx.x;
    if (i < n) dst[i] = (_Float16)src[i];
}

__global__ void zero_f32(float* __restrict__ p, size_t n) {
    size_t i = (size_t)blockIdx.x * blockDim.x + threadIdx.x;
    if (i < n) p[i] = 0.f;
}

// ---------------- conv1x1 GEMM: D = W(MxK) @ X(KxN) --------------------------
// Block 256 = 8 waves. Block tile: 64(M) x 128(N). B panel (32x128 f16, 8KB)
// staged into LDS by the Tensor Data Mover, double-buffered; each wave owns a
// 16-wide N slice and computes 4 m-tiles, so one B operand feeds 4 WMMAs.

template <typename OutT>
__global__ void conv_gemm(const _Float16* __restrict__ A,
                          const _Float16* __restrict__ Bbase,
                          OutT* __restrict__ Dbase, int M, int K) {
    __shared__ __align__(128) _Float16 bpanel[2][32 * 128];
    int tid  = threadIdx.x;
    int lane = tid & 31;
    int wv   = tid >> 5;
    int nBlock = blockIdx.x * 128;
    int mBlock = blockIdx.y * 64;
    int b      = blockIdx.z;
    const _Float16* B = Bbase + (size_t)b * (size_t)K * NTOK;
    OutT*           D = Dbase + (size_t)b * (size_t)M * NTOK;
    int nLocal = wv * 16;
    int half = lane >> 4, l = lane & 15;
    const int KC = K / 32;

    v8f acc[4] = {{}, {}, {}, {}};

#if HAVE_TDM
    unsigned int lds[2] = { (unsigned int)(uintptr_t)&bpanel[0][0],
                            (unsigned int)(uintptr_t)&bpanel[1][0] };
    if (tid == 0)
        tdm_load_2d(lds[0], B + nBlock, 128, 32, NTOK, NTOK, K);
    for (int kc = 0; kc < KC; ++kc) {
        if (tid == 0) {
            if (kc + 1 < KC) {
                tdm_load_2d(lds[(kc + 1) & 1],
                            B + (size_t)(kc + 1) * 32 * NTOK + nBlock,
                            128, 32, NTOK, NTOK, K);
                __builtin_amdgcn_s_wait_tensorcnt(1);   // panel kc is complete
            } else {
                __builtin_amdgcn_s_wait_tensorcnt(0);
            }
        }
        __syncthreads();
        v16h bt = load_b_tile_lds(bpanel[kc & 1], 128, nLocal, lane);
        int kBase = kc * 32;
#pragma unroll
        for (int mi = 0; mi < 4; ++mi) {
            v16h a = load_a_tile(A, K, mBlock + mi * 16, kBase, lane);
            acc[mi] = wmma_f32(a, bt, acc[mi]);
        }
        __syncthreads();   // safe to overwrite this buffer next round
    }
#else
    for (int kc = 0; kc < KC; ++kc) {
        // cooperative 8KB copy: thread t -> row t/8, 16 halves at (t%8)*16
        {
            int row = tid >> 3;
            int cH  = (tid & 7) * 16;
            const uint4* s = (const uint4*)(B + (size_t)(kc * 32 + row) * NTOK +
                                            nBlock + cH);
            uint4* d = (uint4*)&bpanel[0][row * 128 + cH];
            d[0] = s[0];
            d[1] = s[1];
        }
        __syncthreads();
        v16h bt = load_b_tile_lds(bpanel[0], 128, nLocal, lane);
        int kBase = kc * 32;
#pragma unroll
        for (int mi = 0; mi < 4; ++mi) {
            v16h a = load_a_tile(A, K, mBlock + mi * 16, kBase, lane);
            acc[mi] = wmma_f32(a, bt, acc[mi]);
        }
        __syncthreads();
    }
#endif

    int n = nBlock + nLocal + l;
#pragma unroll
    for (int mi = 0; mi < 4; ++mi) {
        V8F r; r.v = acc[mi];
#pragma unroll
        for (int t = 0; t < 8; ++t) {
            int m = mBlock + mi * 16 + half * 8 + t;
            D[(size_t)m * NTOK + n] = (OutT)r.f[t];
        }
    }
}

// ---------------- depthwise 3x3, pad=1 ---------------------------------------

__global__ void dwconv3x3(const _Float16* __restrict__ in, const float* __restrict__ wdw,
                          _Float16* __restrict__ out) {
    size_t t = (size_t)blockIdx.x * blockDim.x + threadIdx.x;  // over B*C3*NTOK
    int x  = (int)(t & (HDIM - 1));
    int y  = (int)((t >> 7) & (HDIM - 1));
    size_t bc = t >> 14;                 // b*C3 + ch
    int ch = (int)(bc % C3);
    const _Float16* p  = in + (bc << 14);
    const float*    wk = wdw + ch * 9;
    float acc = 0.f;
#pragma unroll
    for (int dy = -1; dy <= 1; ++dy)
#pragma unroll
        for (int dx = -1; dx <= 1; ++dx) {
            int yy = y + dy, xx = x + dx;
            if (yy >= 0 && yy < HDIM && xx >= 0 && xx < HDIM)
                acc += (float)p[yy * HDIM + xx] * wk[(dy + 1) * 3 + (dx + 1)];
        }
    out[t] = (_Float16)acc;
}

// ---------------- 1/max(||row||, eps) for q,k rows ---------------------------

__global__ void row_invnorm(const _Float16* __restrict__ QKV, float* __restrict__ invn) {
    int r = blockIdx.x;                       // 0 .. BATCH*384-1
    int b = r / (2 * C_IN), c = r % (2 * C_IN);
    const _Float16* row = QKV + ((size_t)b * C3 + c) * NTOK;
    float ss = 0.f;
    for (int i = threadIdx.x * 8; i < NTOK; i += blockDim.x * 8) {
        V8H u; u.v = *(const v8h*)(row + i);
#pragma unroll
        for (int e = 0; e < 8; ++e) {
            float v = (float)u.h[e];
            ss += v * v;
        }
    }
    __shared__ float red[256];
    red[threadIdx.x] = ss;
    __syncthreads();
    for (int s = 128; s > 0; s >>= 1) {
        if (threadIdx.x < s) red[threadIdx.x] += red[threadIdx.x + s];
        __syncthreads();
    }
    if (threadIdx.x == 0) invn[r] = 1.f / fmaxf(sqrtf(red[0]), 1e-12f);
}

// ---------------- S_raw += q·k^T (split-K, f32 atomics) ----------------------
// grid: (KSPLIT, 3 n-tiles, 16 bh), block 32. One B load feeds 3 WMMAs.

__global__ void attn_gemm(const _Float16* __restrict__ QKV, float* __restrict__ Sraw) {
    int lane = threadIdx.x & 31;
    int ks = blockIdx.x, tj = blockIdx.y, bh = blockIdx.z;
    int b = bh >> 2, h = bh & 3;
    const _Float16* Q  = QKV + ((size_t)b * C3 +        h * CH) * NTOK;
    const _Float16* Km = QKV + ((size_t)b * C3 + C_IN + h * CH) * NTOK;
    int nBase = tj * 16;

    v8f acc[3] = {{}, {}, {}};
    int kSpan = NTOK / KSPLIT;
    int k0 = ks * kSpan;
    for (int kBase = k0; kBase < k0 + kSpan; kBase += 32) {
        v16h bt = load_b_tile_rows(Km, NTOK, nBase, kBase, lane);
#pragma unroll
        for (int mi = 0; mi < 3; ++mi) {
            v16h a = load_a_tile(Q, NTOK, mi * 16, kBase, lane);
            acc[mi] = wmma_f32(a, bt, acc[mi]);
        }
    }
    int half = lane >> 4, l = lane & 15;
    int n = nBase + l;
    float* Sb = Sraw + (size_t)bh * CH * CH;
#pragma unroll
    for (int mi = 0; mi < 3; ++mi) {
        V8F r; r.v = acc[mi];
#pragma unroll
        for (int t = 0; t < 8; ++t) {
            int m = mi * 16 + half * 8 + t;
            atomicAdd(&Sb[m * CH + n], r.f[t]);
        }
    }
}

// ---------------- fold norms * temperature, softmax --------------------------
// Output P padded [48][64] f16, cols 48..63 = 0.

__global__ void softmax48(const float* __restrict__ Sraw, const float* __restrict__ invn,
                          const float* __restrict__ temp, _Float16* __restrict__ P) {
    int bh = blockIdx.x;
    int b = bh >> 2, h = bh & 3;
    int i = threadIdx.x;
    if (i >= CH) return;
    const float* Sb = Sraw + (size_t)bh * CH * CH;
    float invq = invn[b * 2 * C_IN + h * CH + i];
    float T    = temp[h];
    float lg[CH];
    float mx = -1e30f;
    for (int j = 0; j < CH; ++j) {
        float lv = Sb[i * CH + j] * invq * invn[b * 2 * C_IN + C_IN + h * CH + j] * T;
        lg[j] = lv;
        mx = fmaxf(mx, lv);
    }
    float sum = 0.f;
    for (int j = 0; j < CH; ++j) { lg[j] = expf(lg[j] - mx); sum += lg[j]; }
    float inv = 1.f / sum;
    _Float16* Pr = P + (size_t)bh * CH * 64 + i * 64;
    for (int j = 0; j < CH; ++j) Pr[j] = (_Float16)(lg[j] * inv);
    for (int j = CH; j < 64; ++j) Pr[j] = (_Float16)0.f;
}

// ---------------- O = P(48x48, K padded to 64) @ V(48xN) ---------------------
// grid: (NTOK/128, 16 bh), block 256 (8 waves). The v panel [64 k][128 n]
// (rows 48..63 zeroed) is staged once into LDS (TDM when available), then
// each wave runs 2 K-chunks x 3 m-tiles.

__global__ void out_gemm(const _Float16* __restrict__ P, const _Float16* __restrict__ QKV,
                         _Float16* __restrict__ O) {
    __shared__ __align__(128) _Float16 vp[64 * 128];
    int tid  = threadIdx.x;
    int lane = tid & 31;
    int wv   = tid >> 5;
    int nBlock = blockIdx.x * 128;
    int bh = blockIdx.y;
    int b = bh >> 2, h = bh & 3;
    const _Float16* A = P + (size_t)bh * CH * 64;
    const _Float16* V = QKV + ((size_t)b * C3 + 2 * C_IN + h * CH) * NTOK;

#if HAVE_TDM
    if (tid == 0) {
        tdm_load_2d((unsigned int)(uintptr_t)&vp[0], V + nBlock, 128, CH, NTOK,
                    NTOK, CH);
        __builtin_amdgcn_s_wait_tensorcnt(0);
    }
    {   // zero pad rows 48..63 (disjoint from TDM target rows)
        v8h z = {};
        *(v8h*)&vp[CH * 128 + tid * 8] = z;
    }
#else
    for (int r = tid >> 3; r < 64; r += 32) {
        int cH = (tid & 7) * 16;
        uint4* d = (uint4*)&vp[r * 128 + cH];
        if (r < CH) {
            const uint4* s = (const uint4*)(V + (size_t)r * NTOK + nBlock + cH);
            d[0] = s[0];
            d[1] = s[1];
        } else {
            uint4 zz = {0, 0, 0, 0};
            d[0] = zz;
            d[1] = zz;
        }
    }
#endif
    __syncthreads();

    int nLocal = wv * 16;
    int half = lane >> 4, l = lane & 15;
    v8f acc[3] = {{}, {}, {}};
#pragma unroll
    for (int kc = 0; kc < 2; ++kc) {
        v16h bt = load_b_tile_lds(vp + kc * 32 * 128, 128, nLocal, lane);
#pragma unroll
        for (int mi = 0; mi < 3; ++mi) {
            v16h a = load_a_tile(A, 64, mi * 16, kc * 32, lane);
            acc[mi] = wmma_f32(a, bt, acc[mi]);
        }
    }
    _Float16* Ob = O + ((size_t)b * C_IN + h * CH) * NTOK;
    int n = nBlock + nLocal + l;
#pragma unroll
    for (int mi = 0; mi < 3; ++mi) {
        V8F r; r.v = acc[mi];
#pragma unroll
        for (int t = 0; t < 8; ++t) {
            int m = mi * 16 + half * 8 + t;
            Ob[(size_t)m * NTOK + n] = (_Float16)r.f[t];
        }
    }
}

// ---------------- host launch ----------------

extern "C" void kernel_launch(void* const* d_in, const int* in_sizes, int n_in,
                              void* d_out, int out_size, void* d_ws, size_t ws_size,
                              hipStream_t stream) {
    const float* x      = (const float*)d_in[0];   // [4,192,128,128]
    const float* w_qkv  = (const float*)d_in[1];   // [576,192,1,1]
    const float* w_dw   = (const float*)d_in[2];   // [576,1,3,3]
    const float* w_proj = (const float*)d_in[3];   // [192,192,1,1]
    const float* temp   = (const float*)d_in[4];   // [4,1,1]
    float*       out    = (float*)d_out;           // [4,192,128,128]

    char* ws = (char*)d_ws;
    const size_t nX   = (size_t)BATCH * C_IN * NTOK;   // 12,582,912
    const size_t nQKV = (size_t)BATCH * C3 * NTOK;     // 37,748,736
    size_t off = 0;
    _Float16* Xh     = (_Float16*)(ws + off); off += nX * 2;     // reused as O
    _Float16* QKVpre = (_Float16*)(ws + off); off += nQKV * 2;
    _Float16* QKVd   = (_Float16*)(ws + off); off += nQKV * 2;
    _Float16* Wqkvh  = (_Float16*)(ws + off); off += (size_t)C3 * C_IN * 2;
    _Float16* Wprojh = (_Float16*)(ws + off); off += (size_t)C_IN * C_IN * 2;
    float*    invn   = (float*)(ws + off);    off += (size_t)BATCH * 2 * C_IN * 4;
    off = (off + 255) & ~(size_t)255;
    float*    Sraw   = (float*)(ws + off);    off += (size_t)BATCH * HEADS * CH * CH * 4;
    _Float16* P      = (_Float16*)(ws + off); off += (size_t)BATCH * HEADS * CH * 64 * 2;
    _Float16* O      = Xh;  // x-f16 region is dead after stage 1

    // stage 0: f32 -> f16 conversions + zero split-K accumulator
    cvt_f32_to_f16<<<(unsigned)((nX + 255) / 256), 256, 0, stream>>>(x, Xh, nX);
    cvt_f32_to_f16<<<(C3 * C_IN + 255) / 256, 256, 0, stream>>>(w_qkv, Wqkvh,
                                                                (size_t)C3 * C_IN);
    cvt_f32_to_f16<<<(C_IN * C_IN + 255) / 256, 256, 0, stream>>>(w_proj, Wprojh,
                                                                  (size_t)C_IN * C_IN);
    zero_f32<<<(BATCH * HEADS * CH * CH + 255) / 256, 256, 0, stream>>>(
        Sraw, (size_t)BATCH * HEADS * CH * CH);

    // stage 1: qkv GEMM (WMMA, TDM-staged B panels)
    conv_gemm<_Float16><<<dim3(NTOK / 128, C3 / 64, BATCH), 256, 0, stream>>>(
        Wqkvh, Xh, QKVpre, C3, C_IN);

    // stage 2: depthwise 3x3
    dwconv3x3<<<(unsigned)(nQKV / 256), 256, 0, stream>>>(QKVpre, w_dw, QKVd);

    // stage 3: inverse L2 norms for q,k rows
    row_invnorm<<<BATCH * 2 * C_IN, 256, 0, stream>>>(QKVd, invn);

    // stage 4: raw q·k^T with 8-way split-K (WMMA + f32 atomics)
    attn_gemm<<<dim3(KSPLIT, 3, BATCH * HEADS), 32, 0, stream>>>(QKVd, Sraw);

    // stage 5: fold norms + temperature, softmax, pad K 48->64
    softmax48<<<BATCH * HEADS, 64, 0, stream>>>(Sraw, invn, temp, P);

    // stage 6: attn @ v (WMMA, LDS-staged v panel)
    out_gemm<<<dim3(NTOK / 128, BATCH * HEADS), 256, 0, stream>>>(P, QKVd, O);

    // stage 7: output projection GEMM (WMMA, TDM-staged B panels, f32 out)
    conv_gemm<float><<<dim3(NTOK / 128, C_IN / 64, BATCH), 256, 0, stream>>>(
        Wprojh, O, out, C_IN, C_IN);
}
